// GQAAttention_50903952392558
// MI455X (gfx1250) — matrix operand order repaired and checked by reading, not compiled, not run on hardware
//
#include <hip/hip_runtime.h>

// ---------------- problem constants ----------------
constexpr int NB   = 2;      // batch
constexpr int SEQ  = 2048;   // sequence length
constexpr int NH   = 16;     // query heads
constexpr int NKV  = 4;      // kv heads
constexpr int HD   = 64;     // head dim
constexpr int HID  = 1024;   // hidden
constexpr int MTOK = NB * SEQ;          // 4096 tokens
constexpr int NKVD = NKV * HD;          // 256
constexpr int KT   = 64;                // keys per staged LDS tile
constexpr int NT   = SEQ / KT;          // 32 tiles

typedef __bf16 bf16;
typedef __attribute__((ext_vector_type(16))) __bf16 v16bf;
typedef __attribute__((ext_vector_type(8)))  float  v8f;

union FragU { v16bf v; uint4 q[2]; };

// CDNA5 async global->LDS copy (ASYNCcnt-tracked, no VGPR staging).
// VDST = VGPR with LDS byte address; VADDR = 64-bit global address (GV mode).
#define ASYNC_LDS_B128(ldsoff, gptr)                                     \
  asm volatile("global_load_async_to_lds_b128 %0, %1, off"               \
               :: "v"(ldsoff), "v"((unsigned long long)(uintptr_t)(gptr)) \
               : "memory")
#define WAIT_ASYNC(n) asm volatile("s_wait_asynccnt %0" :: "i"(n) : "memory")

// Generic LDS pointer -> wave-relative LDS byte address (aperture keeps the
// LDS offset in addr[31:0] per the ISA flat-addressing rules).
static __device__ __forceinline__ unsigned lds_addr(const void* p) {
  return (unsigned)(uintptr_t)p;
}

// A-matrix fragment (16x32 bf16): lane row = lane%16, lane half h = lane/16.
// Elements 0..15 = K [k0+16h, +16).
static __device__ __forceinline__ v16bf load_a_frag(const bf16* row, int k0, int h) {
  FragU f;
  f.q[0] = *reinterpret_cast<const uint4*>(row + k0 + 16 * h);
  f.q[1] = *reinterpret_cast<const uint4*>(row + k0 + 16 * h + 8);
  return f.v;
}

// B-matrix fragment (32x16 bf16): lane col = lane%16, elements = K [16h, +16).
static __device__ __forceinline__ v16bf load_b_frag(const bf16* p) {
  FragU f;
  f.q[0] = *reinterpret_cast<const uint4*>(p);
  f.q[1] = *reinterpret_cast<const uint4*>(p + 8);
  return f.v;
}

static __device__ __forceinline__ v8f wmma_bf16(v16bf a, v16bf b, v8f c) {
  return __builtin_amdgcn_wmma_f32_16x16x32_bf16(
      /*neg_a=*/false, a, /*neg_b=*/false, b,
      /*c_mod=*/(short)0, c, /*reuse_a=*/false, /*reuse_b=*/false);
}

// ---------------- helper kernels ----------------
__global__ void cast_bf16_kernel(const float* __restrict__ src,
                                 bf16* __restrict__ dst, int n) {
  int i = blockIdx.x * blockDim.x + threadIdx.x;
  if (i < n) dst[i] = (bf16)src[i];
}

// W[K][N] f32 -> WT[N][K] bf16 (contiguous writes)
__global__ void transpose_bf16_kernel(const float* __restrict__ W,
                                      bf16* __restrict__ WT, int K, int N) {
  int i = blockIdx.x * blockDim.x + threadIdx.x;
  if (i < K * N) {
    int k = i % K, n = i / K;
    WT[i] = (bf16)W[(size_t)k * N + n];
  }
}

// ---------------- fused projection GEMM (WMMA bf16) ----------------
enum { MODE_Q = 0, MODE_K = 1, MODE_V = 2, MODE_O = 3 };

template <int MODE>
__global__ __launch_bounds__(128) void proj_kernel(
    const bf16* __restrict__ A, const bf16* __restrict__ WT,
    const float* __restrict__ bias, const int* __restrict__ pos_ids,
    bf16* __restrict__ out_bf, float* __restrict__ out_f32) {
  const int lane = threadIdx.x & 31;
  const int wave = threadIdx.x >> 5;
  const int lr = lane & 15;
  const int h  = lane >> 4;
  const int rowbase = blockIdx.x * 64 + wave * 16;
  const int headi = blockIdx.y;
  const int ncolbase = headi * 64;

  const bf16* arow = A + (size_t)(rowbase + lr) * HID;
  v8f c[4] = {};

  for (int k0 = 0; k0 < HID; k0 += 32) {
    // Hoist all loads ahead of the WMMAs: one load clause + one wait,
    // instead of load/wait/wmma lockstep.
    v16bf a = load_a_frag(arow, k0, h);
    v16bf b[4];
#pragma unroll
    for (int t = 0; t < 4; ++t)
      b[t] = load_b_frag(WT + (size_t)(ncolbase + 16 * t + lr) * HID + k0 + 16 * h);
#pragma unroll
    for (int t = 0; t < 4; ++t) c[t] = wmma_bf16(a, b[t], c[t]);
  }

#pragma unroll
  for (int t = 0; t < 4; ++t) {
    float bv = bias[ncolbase + 16 * t + lr];
#pragma unroll
    for (int r = 0; r < 8; ++r) c[t][r] += bv;
  }

  // RoPE: pair (d, d+32) = C-tiles (t, t+2), same lane.
  if (MODE == MODE_Q || MODE == MODE_K) {
#pragma unroll
    for (int t = 0; t < 2; ++t) {
      const float d32 = (float)(16 * t + lr);
      const float invf = __expf(-d32 * (9.210340371976184f / 32.0f));
#pragma unroll
      for (int r = 0; r < 8; ++r) {
        int row = rowbase + r + 8 * h;
        int bb = row >> 11, s = row & (SEQ - 1);
        float pos = (float)pos_ids[bb * SEQ + s];
        float ang = pos * invf;
        float cs = __cosf(ang), sn = __sinf(ang);
        float q1 = c[t][r], q2 = c[t + 2][r];
        c[t][r]     = q1 * cs - q2 * sn;
        c[t + 2][r] = q2 * cs + q1 * sn;
      }
    }
  }
  if (MODE == MODE_Q) {
#pragma unroll
    for (int t = 0; t < 4; ++t)
#pragma unroll
      for (int r = 0; r < 8; ++r) c[t][r] *= 0.125f;  // fold 1/sqrt(HD)
  }

#pragma unroll
  for (int t = 0; t < 4; ++t) {
#pragma unroll
    for (int r = 0; r < 8; ++r) {
      int row = rowbase + r + 8 * h;
      int col = ncolbase + 16 * t + lr;
      if (MODE == MODE_O) {
        out_f32[(size_t)row * HID + col] = c[t][r];
      } else {
        int bb = row >> 11, s = row & (SEQ - 1);
        int d = 16 * t + lr;
        size_t addr;
        if (MODE == MODE_Q)      addr = ((size_t)(bb * NH  + headi) * SEQ + s) * HD + d;
        else if (MODE == MODE_K) addr = ((size_t)(bb * NKV + headi) * SEQ + s) * HD + d;
        else /* MODE_V */        addr = ((size_t)(bb * NKV + headi) * HD  + d) * SEQ + s;
        out_bf[addr] = (bf16)c[t][r];
      }
    }
  }
}

// ---------------- flash attention: async-LDS staged K/V ----------------
// Block = 4 waves x 16 queries, one head. K/V tiles (64 keys) are DMA'd into
// LDS once per block via global_load_async_to_lds_b128 (double-buffered,
// ASYNCcnt-pipelined); waves read WMMA fragments from LDS (ds_load_b128).
__global__ __launch_bounds__(128) void attn_kernel(
    const bf16* __restrict__ Q,   // [B, NH, S, HD], pre-scaled by 1/8
    const bf16* __restrict__ K,   // [B, NKV, S, HD]
    const bf16* __restrict__ Vt,  // [B, NKV, HD, S]
    bf16* __restrict__ O) {       // [B*S, NH*HD]
  __shared__ alignas(16) bf16 smK[2][KT * HD];  // [key_local][d]
  __shared__ alignas(16) bf16 smV[2][HD * KT];  // [d][key_local]

  const int tid  = threadIdx.x;
  const int lane = tid & 31;
  const int wave = tid >> 5;
  const int lr = lane & 15;
  const int h  = lane >> 4;
  const int q0 = blockIdx.x * 64 + wave * 16;
  const int head = blockIdx.y;
  const int b    = blockIdx.z;
  const int kv   = head >> 2;  // NH/NKV == 4

  const bf16* qrow = Q + ((size_t)(b * NH + head) * SEQ + (q0 + lr)) * HD;
  v16bf bq[2];
#pragma unroll
  for (int f = 0; f < 2; ++f) bq[f] = load_b_frag(qrow + 32 * f + 16 * h);

  const bf16* Kb = K  + (size_t)(b * NKV + kv) * SEQ * HD;
  const bf16* Vb = Vt + (size_t)(b * NKV + kv) * HD * SEQ;

  // Async-stage one 64-key tile: K tile = 8KB contiguous, V tile = 64 rows of
  // 128B. 128 threads x 4 x b128 each per matrix (8 async ops / thread).
  auto issue_tile = [&](int tile, int bufi) {
    const bf16* gK = Kb + (size_t)tile * KT * HD;
#pragma unroll
    for (int j = 0; j < 4; ++j) {
      int ck = tid + 128 * j;  // 16B chunk index, 0..511
      ASYNC_LDS_B128(lds_addr(&smK[bufi][ck * 8]), gK + ck * 8);
    }
#pragma unroll
    for (int j = 0; j < 4; ++j) {
      int cv = tid + 128 * j;
      int dd = cv >> 3, cc = cv & 7;
      ASYNC_LDS_B128(lds_addr(&smV[bufi][dd * KT + cc * 8]),
                     Vb + (size_t)dd * SEQ + tile * KT + cc * 8);
    }
  };

  v8f o[4] = {};
  float mrow = -__builtin_inff();
  float lsum = 0.0f;

  issue_tile(0, 0);
  for (int it = 0; it < NT; ++it) {
    if (it + 1 < NT) {
      issue_tile(it + 1, (it + 1) & 1);
      WAIT_ASYNC(8);   // in-order completion: 8 newest = tile it+1 => tile it done
    } else {
      WAIT_ASYNC(0);
    }
    __syncthreads();   // all waves' shares of tile `it` visible
    const bf16* bK = smK[it & 1];
    const bf16* bV = smV[it & 1];

#pragma unroll
    for (int kk = 0; kk < 2; ++kk) {  // two 32-key halves of the tile
      v8f s0 = {}, s1 = {};
#pragma unroll
      for (int f = 0; f < 2; ++f) {
        v16bf a0 = load_a_frag(bK + (32 * kk + lr) * HD, 32 * f, h);       // ds_load
        v16bf a1 = load_a_frag(bK + (32 * kk + 16 + lr) * HD, 32 * f, h);
        s0 = wmma_bf16(a0, bq[f], s0);
        s1 = wmma_bf16(a1, bq[f], s1);
      }
      // online softmax; lane stat is for query column lr
      float tm = -__builtin_inff();
#pragma unroll
      for (int r = 0; r < 8; ++r) tm = fmaxf(tm, fmaxf(s0[r], s1[r]));
      tm = fmaxf(tm, __shfl_xor(tm, 16, 32));
      float nm = fmaxf(mrow, tm);
      float alpha = __expf(mrow - nm);
      float p0[8], p1[8], ps = 0.0f;
#pragma unroll
      for (int r = 0; r < 8; ++r) {
        p0[r] = __expf(s0[r] - nm);
        p1[r] = __expf(s1[r] - nm);
        ps += p0[r] + p1[r];
      }
      ps += __shfl_xor(ps, 16, 32);
      lsum = lsum * alpha + ps;
      mrow = nm;
#pragma unroll
      for (int dt = 0; dt < 4; ++dt)
#pragma unroll
        for (int r = 0; r < 8; ++r) o[dt][r] *= alpha;

      // Build P^T B-fragment: swap the 8 other-half key values with lane^16.
      float recv[8];
#pragma unroll
      for (int r = 0; r < 8; ++r) {
        float send = h ? p0[r] : p1[r];
        recv[r] = __shfl_xor(send, 16, 32);
      }
      v16bf pb;
#pragma unroll
      for (int e = 0; e < 8; ++e) {
        float lo = h ? recv[e] : p0[e];  // keys [16h, 16h+8)
        float hi = h ? p1[e]   : recv[e];
        pb[e]     = (bf16)lo;
        pb[e + 8] = (bf16)hi;
      }
      // O^T += V^T . P^T
#pragma unroll
      for (int dt = 0; dt < 4; ++dt) {
        v16bf av = load_a_frag(bV + (16 * dt + lr) * KT, 32 * kk, h);      // ds_load
        o[dt] = wmma_bf16(av, pb, o[dt]);
      }
    }
    __syncthreads();   // done reading buf[it&1] before it is re-issued
  }

  float inv = 1.0f / lsum;
  const int tok = b * SEQ + q0 + lr;
#pragma unroll
  for (int dt = 0; dt < 4; ++dt)
#pragma unroll
    for (int r = 0; r < 8; ++r) {
      int d = 16 * dt + r + 8 * h;
      O[(size_t)tok * HID + head * HD + d] = (bf16)(o[dt][r] * inv);
    }
}

// ---------------- host launch ----------------
extern "C" void kernel_launch(void* const* d_in, const int* in_sizes, int n_in,
                              void* d_out, int out_size, void* d_ws, size_t ws_size,
                              hipStream_t stream) {
  const float* x  = (const float*)d_in[0];
  const int*  pos = (const int*)d_in[1];
  const float* Wq = (const float*)d_in[2];
  const float* bq = (const float*)d_in[3];
  const float* Wk = (const float*)d_in[4];
  const float* bk = (const float*)d_in[5];
  const float* Wv = (const float*)d_in[6];
  const float* bv = (const float*)d_in[7];
  const float* Wo = (const float*)d_in[8];
  const float* bo = (const float*)d_in[9];
  float* out = (float*)d_out;

  char* ws = (char*)d_ws;
  size_t off = 0;
  auto alloc = [&](size_t bytes) {
    char* p = ws + off;
    off += (bytes + 255) & ~size_t(255);
    return p;
  };
  bf16* xbf = (bf16*)alloc((size_t)MTOK * HID * 2);
  bf16* wtq = (bf16*)alloc((size_t)HID * HID * 2);
  bf16* wtk = (bf16*)alloc((size_t)NKVD * HID * 2);
  bf16* wtv = (bf16*)alloc((size_t)NKVD * HID * 2);
  bf16* wto = (bf16*)alloc((size_t)HID * HID * 2);
  bf16* qbf = (bf16*)alloc((size_t)NB * NH * SEQ * HD * 2);
  bf16* kbf = (bf16*)alloc((size_t)NB * NKV * SEQ * HD * 2);
  bf16* vtb = (bf16*)alloc((size_t)NB * NKV * HD * SEQ * 2);
  bf16* att = (bf16*)alloc((size_t)MTOK * HID * 2);

  const int nx = MTOK * HID;
  cast_bf16_kernel<<<(nx + 255) / 256, 256, 0, stream>>>(x, xbf, nx);
  transpose_bf16_kernel<<<(HID * HID + 255) / 256, 256, 0, stream>>>(Wq, wtq, HID, HID);
  transpose_bf16_kernel<<<(HID * NKVD + 255) / 256, 256, 0, stream>>>(Wk, wtk, HID, NKVD);
  transpose_bf16_kernel<<<(HID * NKVD + 255) / 256, 256, 0, stream>>>(Wv, wtv, HID, NKVD);
  transpose_bf16_kernel<<<(HID * HID + 255) / 256, 256, 0, stream>>>(Wo, wto, HID, HID);

  proj_kernel<MODE_Q><<<dim3(MTOK / 64, NH),  128, 0, stream>>>(xbf, wtq, bq, pos, qbf, nullptr);
  proj_kernel<MODE_K><<<dim3(MTOK / 64, NKV), 128, 0, stream>>>(xbf, wtk, bk, pos, kbf, nullptr);
  proj_kernel<MODE_V><<<dim3(MTOK / 64, NKV), 128, 0, stream>>>(xbf, wtv, bv, pos, vtb, nullptr);

  attn_kernel<<<dim3(SEQ / 64, NH, NB), 128, 0, stream>>>(qbf, kbf, vtb, att);

  proj_kernel<MODE_O><<<dim3(MTOK / 64, NH), 128, 0, stream>>>(att, wto, bo, pos, nullptr, out);
}
